// DisLoss_77309412135
// MI455X (gfx1250) — matrix operand order repaired and checked by reading, not compiled, not run on hardware
//
#include <hip/hip_runtime.h>
#include <math.h>

// ---------------------------------------------------------------------------
// DisLoss for MI455X (gfx1250, wave32).
// Problem sizes (fixed by the reference): B=4096 rows per chunk, D=2048,
// K_SIZE=8 -> 512 groups.  Memory floor ~100MB @ 23.3 TB/s ~= 4.3us.
// The 512x512x2048 center Gram matrix (~1.07 GFLOP) is the only dense-compute
// piece; it runs on v_wmma_f32_16x16x32_f16 with an f16 hi/lo split of the
// centers (3 WMMAs per K=32 step) for ~f32 accuracy.
// ---------------------------------------------------------------------------

typedef __attribute__((ext_vector_type(4)))  float    v4f;
typedef __attribute__((ext_vector_type(8)))  float    v8f;
typedef __attribute__((ext_vector_type(8)))  _Float16 v8h;
typedef __attribute__((ext_vector_type(16))) _Float16 v16h;

#define BB 4096
#define DD 2048
#define NG 512          // number of groups
#define D4 (DD / 4)     // 512 float4 per row

// ---------------------------------------------------------------------------
// Kernel 1: group sums.  One thread owns (group g, 4 columns).
//   cm[g]  = mean of 8 fm rows      (fm = inputs rows [2B, 3B))
//   c[g]   = mean of 16 f1/f2 rows  (rows [0,B) and [B,2B))
// Also emits the f16 hi/lo split of c for the WMMA kernel.
// ---------------------------------------------------------------------------
__global__ void k_group_sums(const float* __restrict__ inp,
                             float* __restrict__ cmat,
                             float* __restrict__ cm,
                             _Float16* __restrict__ chi,
                             _Float16* __restrict__ clo) {
    int gid = blockIdx.x * blockDim.x + threadIdx.x;   // 0 .. NG*D4-1
    int g   = gid >> 9;        // group index
    int c4  = gid & 511;       // float4 column index
    const v4f* in4 = (const v4f*)inp;

    v4f sm = {};
#pragma unroll
    for (int r = 0; r < 8; ++r)
        sm += in4[(size_t)(2 * BB + g * 8 + r) * D4 + c4];

    v4f sf = {};
#pragma unroll
    for (int r = 0; r < 8; ++r)
        sf += in4[(size_t)(g * 8 + r) * D4 + c4];
#pragma unroll
    for (int r = 0; r < 8; ++r)
        sf += in4[(size_t)(BB + g * 8 + r) * D4 + c4];

    v4f cmv = sm * 0.125f;     // /8
    v4f cv  = sf * 0.0625f;    // /16
    ((v4f*)cm)[(size_t)g * D4 + c4]   = cmv;
    ((v4f*)cmat)[(size_t)g * D4 + c4] = cv;

    int base = g * DD + (c4 << 2);
#pragma unroll
    for (int j = 0; j < 4; ++j) {
        float x = cv[j];
        _Float16 hi = (_Float16)x;
        chi[base + j] = hi;
        clo[base + j] = (_Float16)(x - (float)hi);   // residual (~2^-22 rel)
    }
}

// ---------------------------------------------------------------------------
// Kernel 2: sq[g] = ||c_g||^2  (512 blocks, deterministic LDS tree)
// ---------------------------------------------------------------------------
__global__ void k_rowsq(const float* __restrict__ cmat, float* __restrict__ sq) {
    __shared__ float sh[256];
    int g = blockIdx.x;
    const v4f* r4 = (const v4f*)(cmat + (size_t)g * DD);
    float p = 0.f;
    for (int j = threadIdx.x; j < D4; j += 256) {
        v4f v = r4[j];
        p += v.x * v.x + v.y * v.y + v.z * v.z + v.w * v.w;
    }
    sh[threadIdx.x] = p;
    __syncthreads();
    for (int s = 128; s > 0; s >>= 1) {
        if (threadIdx.x < s) sh[threadIdx.x] += sh[threadIdx.x + s];
        __syncthreads();
    }
    if (threadIdx.x == 0) sq[g] = sh[0];
}

// ---------------------------------------------------------------------------
// Kernel 3: dist_pc[i] = ||row_i - cm[g(i)]||, rows 0..2B-1 (f1 then f2,
// contiguous in `inp`).  One block per row.
// ---------------------------------------------------------------------------
__global__ void k_dist_pc(const float* __restrict__ inp,
                          const float* __restrict__ cm,
                          float* __restrict__ dist_pc) {
    __shared__ float sh[256];
    int row = blockIdx.x;                    // 0 .. 2B-1
    int lab = (row & (BB - 1)) >> 3;
    const v4f* r4 = (const v4f*)(inp + (size_t)row * DD);
    const v4f* c4 = (const v4f*)(cm + (size_t)lab * DD);
    float p = 0.f;
    for (int j = threadIdx.x; j < D4; j += 256) {
        v4f d = r4[j] - c4[j];
        p += d.x * d.x + d.y * d.y + d.z * d.z + d.w * d.w;
    }
    sh[threadIdx.x] = p;
    __syncthreads();
    for (int s = 128; s > 0; s >>= 1) {
        if (threadIdx.x < s) sh[threadIdx.x] += sh[threadIdx.x + s];
        __syncthreads();
    }
    if (threadIdx.x == 0) dist_pc[row] = sqrtf(sh[0]);
}

// ---------------------------------------------------------------------------
// Kernel 4: WMMA center-distance tiles.  1024 tiles of 16x16 over the 512x512
// Gram matrix; one wave32 per tile, 8 waves per block.  A/B operands packed
// in contiguous k order, same for A and B (lane m=l&15, half h=l>>4):
//   A[e]=row[k0+16h+e];  B[e]=col[k0+16h+e].
// Gram = hiB7hi + hiB7lo + loB7hi  (f16 split), f32 accumulate.
// ---------------------------------------------------------------------------
__global__ void k_center_dist(const _Float16* __restrict__ chi,
                              const _Float16* __restrict__ clo,
                              const float* __restrict__ sq,
                              float* __restrict__ tsum) {
    int wave = threadIdx.x >> 5;
    int lane = threadIdx.x & 31;
    int tile = blockIdx.x * 8 + wave;        // 0..1023
    int ti = tile >> 5, tj = tile & 31;
    int ln = lane & 15;
    int h  = lane >> 4;

    const _Float16* aH = chi + (size_t)(ti * 16 + ln) * DD;
    const _Float16* aL = clo + (size_t)(ti * 16 + ln) * DD;
    const _Float16* bH = chi + (size_t)(tj * 16 + ln) * DD;
    const _Float16* bL = clo + (size_t)(tj * 16 + ln) * DD;
    int a0 = 16 * h, a1 = 16 * h + 8, b0 = 16 * h;

    v8f acc = {};
#pragma unroll 4
    for (int k = 0; k < DD; k += 32) {
        v8h ah0 = *(const v8h*)(aH + k + a0);
        v8h ah1 = *(const v8h*)(aH + k + a1);
        v8h al0 = *(const v8h*)(aL + k + a0);
        v8h al1 = *(const v8h*)(aL + k + a1);
        v16h Ah = __builtin_shufflevector(ah0, ah1,
                  0,1,2,3,4,5,6,7,8,9,10,11,12,13,14,15);
        v16h Al = __builtin_shufflevector(al0, al1,
                  0,1,2,3,4,5,6,7,8,9,10,11,12,13,14,15);
        v16h Bh = *(const v16h*)(bH + k + b0);
        v16h Bl = *(const v16h*)(bL + k + b0);
        acc = __builtin_amdgcn_wmma_f32_16x16x32_f16(false, Ah, false, Bh,
                                                     (short)0, acc, false, false);
        acc = __builtin_amdgcn_wmma_f32_16x16x32_f16(false, Ah, false, Bl,
                                                     (short)0, acc, false, false);
        acc = __builtin_amdgcn_wmma_f32_16x16x32_f16(false, Al, false, Bh,
                                                     (short)0, acc, false, false);
    }

    // C/D layout: element r -> M = r + 8h, N = lane&15.
    int gcol = tj * 16 + ln;
    float sqj = sq[gcol];
    const v4f* sqi4 = (const v4f*)(sq + ti * 16 + 8 * h);
    v4f si0 = sqi4[0], si1 = sqi4[1];
    float sqi[8] = { si0.x, si0.y, si0.z, si0.w, si1.x, si1.y, si1.z, si1.w };
    int grow0 = ti * 16 + 8 * h;

    float local = 0.f;
#pragma unroll
    for (int r = 0; r < 8; ++r) {
        float s = sqi[r] + sqj - 2.0f * acc[r];
        float d = sqrtf(fmaxf(s, 1e-12f));
        if (grow0 + r != gcol) local += d;   // exclude same-group (diagonal)
    }
    // deterministic in-wave tree reduction (wave32)
    for (int off = 16; off > 0; off >>= 1)
        local += __shfl_down(local, off, 32);
    if (lane == 0) tsum[tile] = local;
}

// ---------------------------------------------------------------------------
// Kernel 5: final deterministic reduction -> scalar loss.
//   loss = (sum(dist_pc)/4096) / (16*T / (8176*512))
// ---------------------------------------------------------------------------
__global__ void k_final(const float* __restrict__ dist_pc,
                        const float* __restrict__ tsum,
                        float* __restrict__ out) {
    __shared__ float sh[256];
    float p = 0.f;
    for (int i = threadIdx.x; i < 2 * BB; i += 256) p += dist_pc[i];
    sh[threadIdx.x] = p;
    __syncthreads();
    for (int s = 128; s > 0; s >>= 1) {
        if (threadIdx.x < s) sh[threadIdx.x] += sh[threadIdx.x + s];
        __syncthreads();
    }
    float sum_pc = sh[0];
    __syncthreads();

    float t = 0.f;
    for (int i = threadIdx.x; i < 1024; i += 256) t += tsum[i];
    sh[threadIdx.x] = t;
    __syncthreads();
    for (int s = 128; s > 0; s >>= 1) {
        if (threadIdx.x < s) sh[threadIdx.x] += sh[threadIdx.x + s];
        __syncthreads();
    }
    if (threadIdx.x == 0) {
        float T   = sh[0];
        float num = sum_pc / 4096.0f;
        float den = (16.0f * T) / (8176.0f * 512.0f);
        out[0] = num / den;
    }
}

// ---------------------------------------------------------------------------
// Host launcher.  Workspace layout (bytes):
//   [0,      4MB)  c    (512x2048 f32)
//   [4MB,    8MB)  cm   (512x2048 f32)
//   [8MB,   +4KB)  sq   (512 f32)
//   [+4KB, +32KB)  dist_pc (8192 f32)
//   [+36KB, +4KB)  tile sums (1024 f32)
//   [+40KB,  2MB)  c_hi (f16)
//   [....,   2MB)  c_lo (f16)        total ~12.05 MB
// ---------------------------------------------------------------------------
extern "C" void kernel_launch(void* const* d_in, const int* in_sizes, int n_in,
                              void* d_out, int out_size, void* d_ws, size_t ws_size,
                              hipStream_t stream) {
    (void)in_sizes; (void)n_in; (void)out_size; (void)ws_size;
    const float* inp = (const float*)d_in[0];   // (3B, D) f32; targets/k_size implied

    char* ws = (char*)d_ws;
    float*    cmat   = (float*)(ws);
    float*    cm     = (float*)(ws + ((size_t)4 << 20));
    float*    sq     = (float*)(ws + ((size_t)8 << 20));
    float*    distpc = (float*)(ws + ((size_t)8 << 20) + 4096);
    float*    tsum   = (float*)(ws + ((size_t)8 << 20) + 4096 + 32768);
    _Float16* chi    = (_Float16*)(ws + ((size_t)8 << 20) + 4096 + 32768 + 4096);
    _Float16* clo    = (_Float16*)(ws + ((size_t)8 << 20) + 4096 + 32768 + 4096
                                      + ((size_t)2 << 20));

    k_group_sums<<<(NG * D4) / 256, 256, 0, stream>>>(inp, cmat, cm, chi, clo);
    k_rowsq     <<<NG,   256, 0, stream>>>(cmat, sq);
    k_dist_pc   <<<2*BB, 256, 0, stream>>>(inp, cm, distpc);
    k_center_dist<<<128, 256, 0, stream>>>(chi, clo, sq, tsum);
    k_final     <<<1,    256, 0, stream>>>(distpc, tsum, (float*)d_out);
}
